// MimiDecoderONNX_36197984371417
// MI455X (gfx1250) — compile-verified, matrix operand order checked by reading, not run on hardware
//
#include <hip/hip_runtime.h>
#include <math.h>

typedef __attribute__((ext_vector_type(16))) _Float16 v16h;
typedef __attribute__((ext_vector_type(8)))  _Float16 v8h;
typedef __attribute__((ext_vector_type(8)))  float    v8f;
typedef __attribute__((ext_vector_type(4)))  int      v4i;

#define FLAG_BIAS_M  1
#define FLAG_GELU    2
#define FLAG_BELU    4
#define FLAG_BHALF   8
#define FLAG_OUTHALF 16

// CDNA5 async global->LDS copy path (pure f16 weight tiles only).
#if defined(__gfx1250__) && __has_builtin(__builtin_amdgcn_global_load_async_to_lds_b128) && __has_builtin(__builtin_amdgcn_s_wait_asynccnt)
#define ASYNC_LDS 1
typedef __attribute__((address_space(1))) v4i* gv4i_p;
typedef __attribute__((address_space(3))) v4i* lv4i_p;
__device__ __forceinline__ void async_cp16(const void* g, void* l) {
  __builtin_amdgcn_global_load_async_to_lds_b128((gv4i_p)g, (lv4i_p)l, 0, 0);
}
__device__ __forceinline__ void async_wait_all() { __builtin_amdgcn_s_wait_asynccnt(0); }
#else
#define ASYNC_LDS 0
#endif

struct GP {
  const void* A; const void* B; const float* bias; const float* ls; const void* res; void* out;
  int M, N, K, lda, ldb, ks, cout, flags, pstride;
  long long out_sm, out_sn;
};

__device__ __forceinline__ v8h zero8h() {
  v8h h;
#pragma unroll
  for (int i = 0; i < 8; ++i) h[i] = (_Float16)0.0f;
  return h;
}

// Load one 16x32 fragment row/col (32 contiguous halves in LDS) into the
// CDNA5 wave32 16-bit WMMA K-interleave:
// lanes 0-15 hold K 0..7 & 16..23, lanes 16-31 hold K 8..15 & 24..31.
__device__ __forceinline__ v16h frag_ld(const _Float16* row, int lane) {
  const int hi = (lane & 16) ? 8 : 0;
  v8h a = *(const v8h*)(row + hi);
  v8h b = *(const v8h*)(row + 16 + hi);
  return __builtin_shufflevector(a, b, 0,1,2,3,4,5,6,7,8,9,10,11,12,13,14,15);
}

// MODE 0 = LINEAR : A = f32 activations (M=T rows, lda), B = f16 weights (N x K row-major, ldb=K)
// MODE 1 = PLAIN  : A = f16 weights (M x K row-major), B = f32 activations (K x N, row stride ldb)
// MODE 2 = CONV   : A = f16 weights (Cout x Cin*ks), B = activations (Cin x Tin), causal pad ks-1
// MODE 3 = CONVTR : A = f16 weights gathered from (Cin, Cout, 2s), per-phase 2-tap GEMM, K = 2*Cin
// NOTE: every launch uses K % 32 == 0, so no K-edge handling in the stagers.
// Double-buffered LDS pipeline: tile kc+32 is staged (async engine + converts)
// while WMMAs consume tile kc; one s_wait_asynccnt + one barrier per step.
template<int MODE>
__global__ __launch_bounds__(256) void gemm_wmma(GP p) {
  __shared__ __align__(16) _Float16 lA[2][64 * 40];
  __shared__ __align__(16) _Float16 lB[2][128 * 40];
  const int tid  = threadIdx.x;
  const int lane = tid & 31;
  const int wid  = tid >> 5;
  const int wm   = (wid >> 2) * 32;   // wave row base within WG tile (0/32)
  const int wn   = (wid & 3) * 32;    // wave col base within WG tile (0..96)
  const int phase = blockIdx.z;
  const int s2 = p.ks >> 1;           // conv-transpose stride (ks == 2*s)
  v8f acc[2][2] = {};

  auto stage = [&](int kc, _Float16* sA, _Float16* sB) {
    // ---------------- stage A tile: 64 rows x 32 K ----------------
    {
      const int mL  = tid >> 2;
      const int k0q = (tid & 3) * 8;
      const int m   = blockIdx.y * 64 + mL;
      const int k   = kc + k0q;
      _Float16* dst = &sA[mL * 40 + k0q];
      if (MODE == 0) {            // f32 activations, contiguous 8 floats
        v8h h = zero8h();
        if (m < p.M) {
          float f[8];
          __builtin_memcpy(f, (const float*)p.A + (long long)m * p.lda + k, 32);
#pragma unroll
          for (int i = 0; i < 8; ++i) h[i] = (_Float16)f[i];
        }
        *(v8h*)dst = h;
      } else if (MODE == 3) {     // per-phase gathered conv-transpose weights
        v8h h = zero8h();
        if (m < p.M) {
#pragma unroll
          for (int i = 0; i < 8; ++i) {
            const int kk2 = k + i;
            const int cin = kk2 >> 1;
            const int tap = (kk2 & 1) ? phase : (s2 + phase);
            h[i] = ((const _Float16*)p.A)[((long long)cin * p.cout + m) * p.ks + tap];
          }
        }
        *(v8h*)dst = h;
      } else {                    // f16 weights, contiguous 8 halves (16 B)
#if ASYNC_LDS
        if (m < p.M) async_cp16((const _Float16*)p.A + (long long)m * p.K + k, dst);
        // rows m >= M keep stale LDS; they only feed discarded C rows.
#else
        v8h h = zero8h();
        if (m < p.M) __builtin_memcpy(&h, (const _Float16*)p.A + (long long)m * p.K + k, 16);
        *(v8h*)dst = h;
#endif
      }
    }
    // ---------------- stage B tile: 32 K x 128 cols (LDS col-major [n][k]) ----------------
    if (MODE == 0) {              // f16 weights: row n holds 16 contiguous halves (32 B)
      const int nL = tid >> 1;
      const int k0 = (tid & 1) * 16;
      const int n  = blockIdx.x * 128 + nL;
      _Float16* dst = &sB[nL * 40 + k0];
      const _Float16* src = (const _Float16*)p.B + (long long)n * p.ldb + kc + k0;
#if ASYNC_LDS
      if (n < p.N) { async_cp16(src, dst); async_cp16(src + 8, dst + 8); }
#else
      v8h h0 = zero8h(), h1 = zero8h();
      if (n < p.N) { __builtin_memcpy(&h0, src, 16); __builtin_memcpy(&h1, src + 8, 16); }
      *(v8h*)dst = h0; *(v8h*)(dst + 8) = h1;
#endif
    } else {                      // activation source: one K-row, 16 contiguous n per thread
      const int kL = tid >> 3;
      const int n0 = (tid & 7) * 16;
      const int k  = kc + kL;
      const int nBase = blockIdx.x * 128 + n0;
      long long row, t0;
      if (MODE == 1) {
        row = (long long)k * p.ldb; t0 = nBase;
      } else if (MODE == 2) {
        const int cin = k / p.ks;
        const int kk  = k - cin * p.ks;
        row = (long long)cin * p.ldb; t0 = (long long)nBase - (p.ks - 1) + kk;
      } else {
        const int cin = k >> 1;
        row = (long long)cin * p.ldb; t0 = (long long)nBase - 1 + (k & 1);
      }
      const bool belu = (p.flags & FLAG_BELU) != 0;
      const bool bhlf = (p.flags & FLAG_BHALF) != 0;
      const long long tlim = (MODE == 1) ? (long long)p.N : (long long)p.ldb;
      _Float16* dst = &sB[n0 * 40 + kL];
      if (nBase + 16 <= p.N && t0 >= 0 && t0 + 16 <= tlim) {   // whole-run fast path
        float f[16];
        if (bhlf) {
          _Float16 hh[16];
          __builtin_memcpy(hh, (const _Float16*)p.B + row + t0, 32);
#pragma unroll
          for (int i = 0; i < 16; ++i) f[i] = (float)hh[i];
        } else {
          __builtin_memcpy(f, (const float*)p.B + row + t0, 64);
        }
#pragma unroll
        for (int i = 0; i < 16; ++i) {
          float v = f[i];
          if (belu && v < 0.0f) v = expm1f(v);
          dst[i * 40] = (_Float16)v;
        }
      } else {                                                 // boundary tiles only
#pragma unroll
        for (int i = 0; i < 16; ++i) {
          const int n = nBase + i;
          const long long t = t0 + i;
          float v = 0.0f;
          if (n < p.N && t >= 0 && t < tlim) {
            v = bhlf ? (float)((const _Float16*)p.B)[row + t] : ((const float*)p.B)[row + t];
            if (belu && v < 0.0f) v = expm1f(v);
          }
          dst[i * 40] = (_Float16)v;
        }
      }
    }
  };

  // pipeline prologue
  stage(0, lA[0], lB[0]);
#if ASYNC_LDS
  async_wait_all();
#endif
  __syncthreads();

  int buf = 0;
  for (int kc = 0; kc < p.K; kc += 32) {
    if (kc + 32 < p.K) stage(kc + 32, lA[buf ^ 1], lB[buf ^ 1]);  // prefetch next tile
    const _Float16* cA = lA[buf];
    const _Float16* cB = lB[buf];
    v16h aF0 = frag_ld(&cA[(wm +      (lane & 15)) * 40], lane);
    v16h aF1 = frag_ld(&cA[(wm + 16 + (lane & 15)) * 40], lane);
    v16h bF0 = frag_ld(&cB[(wn +      (lane & 15)) * 40], lane);
    v16h bF1 = frag_ld(&cB[(wn + 16 + (lane & 15)) * 40], lane);
    acc[0][0] = __builtin_amdgcn_wmma_f32_16x16x32_f16(false, aF0, false, bF0, (short)0, acc[0][0], false, false);
    acc[0][1] = __builtin_amdgcn_wmma_f32_16x16x32_f16(false, aF0, false, bF1, (short)0, acc[0][1], false, false);
    acc[1][0] = __builtin_amdgcn_wmma_f32_16x16x32_f16(false, aF1, false, bF0, (short)0, acc[1][0], false, false);
    acc[1][1] = __builtin_amdgcn_wmma_f32_16x16x32_f16(false, aF1, false, bF1, (short)0, acc[1][1], false, false);
#if ASYNC_LDS
    async_wait_all();
#endif
    __syncthreads();
    buf ^= 1;
  }

  // Epilogue. C/D layout: VGPR v holds M = v + 8*(lane>=16), N = lane&15.
  const int r0 = (lane >> 4) << 3;
  const int nn = lane & 15;
#pragma unroll
  for (int tm = 0; tm < 2; ++tm)
#pragma unroll
    for (int tn = 0; tn < 2; ++tn) {
      const int n = blockIdx.x * 128 + wn + tn * 16 + nn;
      const int mBase = blockIdx.y * 64 + wm + tm * 16 + r0;
#pragma unroll
      for (int v = 0; v < 8; ++v) {
        const int m = mBase + v;
        if (m < p.M && n < p.N) {
          const long long t = (long long)phase + (long long)p.pstride * n;
          const int ch = (p.flags & FLAG_BIAS_M) ? m : n;
          float val = acc[tm][tn][v];
          if (p.bias) val += p.bias[ch];
          if (p.flags & FLAG_GELU) val = 0.5f * val * (1.0f + erff(val * 0.70710678118f));
          if (p.ls) val *= p.ls[ch];
          const long long off = (long long)m * p.out_sm + t * p.out_sn;
          if (p.res) val += (p.flags & FLAG_OUTHALF) ? (float)((const _Float16*)p.res)[off]
                                                     : ((const float*)p.res)[off];
          if (p.flags & FLAG_OUTHALF) ((_Float16*)p.out)[off] = (_Float16)val;
          else                        ((float*)p.out)[off] = val;
        }
      }
    }
}

__global__ void cvt_f32_f16(const float* __restrict__ s, _Float16* __restrict__ d, int n) {
  int i = blockIdx.x * 256 + threadIdx.x;
  if (i < n) d[i] = (_Float16)s[i];
}

__global__ void embed_k(const float* __restrict__ lat, const float* __restrict__ stdv,
                        const float* __restrict__ meanv, float* __restrict__ o) {
  int i = blockIdx.x * 256 + threadIdx.x;
  if (i < 512 * 512) { int d = i >> 9; o[i] = lat[i] * stdv[d] + meanv[d]; }
}

__global__ void ln_k(const float* __restrict__ x, const float* __restrict__ w,
                     const float* __restrict__ b, float* __restrict__ o, int T) {
  const int lane = threadIdx.x & 31, wid = threadIdx.x >> 5;
  const int row = blockIdx.x * 8 + wid;
  if (row >= T) return;
  const float* xr = x + (long long)row * 512;
  float v[16]; float s = 0.f;
#pragma unroll
  for (int i = 0; i < 16; ++i) { v[i] = xr[i * 32 + lane]; s += v[i]; }
#pragma unroll
  for (int o2 = 16; o2; o2 >>= 1) s += __shfl_xor(s, o2, 32);
  const float mean = s * (1.0f / 512.0f);
  float q = 0.f;
#pragma unroll
  for (int i = 0; i < 16; ++i) { float d = v[i] - mean; q += d * d; }
#pragma unroll
  for (int o2 = 16; o2; o2 >>= 1) q += __shfl_xor(q, o2, 32);
  const float rstd = rsqrtf(q * (1.0f / 512.0f) + 1e-5f);
  float* orow = o + (long long)row * 512;
#pragma unroll
  for (int i = 0; i < 16; ++i) {
    const int d0 = i * 32 + lane;
    orow[d0] = (v[i] - mean) * rstd * w[d0] + b[d0];
  }
}

__global__ void rope_k(float* __restrict__ qkv, int T) {
  int idx = blockIdx.x * 256 + threadIdx.x;
  if (idx >= T * 256) return;
  const int t = idx >> 8, rem = idx & 255, h = rem >> 5, i = rem & 31;
  const float fr = expf(-0.2878231366242557f * (float)i); // -ln(1e4)*2/64
  const float ang = (float)t * fr;
  const float c = cosf(ang), sn = sinf(ang);
  const long long base = (long long)t * 1536 + h * 64 + 2 * i;
  float re = qkv[base], im = qkv[base + 1];
  qkv[base] = re * c - im * sn; qkv[base + 1] = re * sn + im * c;
  re = qkv[base + 512]; im = qkv[base + 513];
  qkv[base + 512] = re * c - im * sn; qkv[base + 513] = re * sn + im * c;
}

// Windowed causal attention: one wave32 per (query, head). CTX=250 keys max.
__global__ __launch_bounds__(256) void attn_k(const float* __restrict__ qkv,
                                              float* __restrict__ o, int T) {
  __shared__ float sp[8][256];
  const int lane = threadIdx.x & 31, wid = threadIdx.x >> 5;
  int qh = blockIdx.x * 8 + wid;
  if (qh >= T * 8) qh = T * 8 - 1;
  const int qt = qh >> 3, h = qh & 7;
  const int kstart = (qt > 249) ? qt - 249 : 0;
  const int nk = qt - kstart + 1;
  const float* qp = qkv + (long long)qt * 1536 + h * 64;
  float smax = -1e30f;
  for (int j = lane; j < nk; j += 32) {
    const float* kp = qkv + (long long)(kstart + j) * 1536 + 512 + h * 64;
    float s = 0.f;
#pragma unroll
    for (int d = 0; d < 64; ++d) s += qp[d] * kp[d];
    s *= 0.125f;
    sp[wid][j] = s;
    smax = fmaxf(smax, s);
  }
#pragma unroll
  for (int o2 = 16; o2; o2 >>= 1) smax = fmaxf(smax, __shfl_xor(smax, o2, 32));
  float ssum = 0.f;
  for (int j = lane; j < nk; j += 32) {
    const float pv = expf(sp[wid][j] - smax);
    sp[wid][j] = pv;
    ssum += pv;
  }
#pragma unroll
  for (int o2 = 16; o2; o2 >>= 1) ssum += __shfl_xor(ssum, o2, 32);
  const float inv = 1.0f / ssum;
  __syncthreads();
  const float* vp = qkv + (long long)kstart * 1536 + 1024 + h * 64;
#pragma unroll
  for (int ds = 0; ds < 2; ++ds) {
    const int d = lane + 32 * ds;
    float a = 0.f;
    for (int j = 0; j < nk; ++j) a += sp[wid][j] * vp[(long long)j * 1536 + d];
    o[(long long)qt * 512 + h * 64 + d] = a * inv;
  }
}

static inline int ceil_div_i(long long a, long long b) { return (int)((a + b - 1) / b); }

static void launch_gemm(hipStream_t st, int mode, GP p, int phases) {
  dim3 g(ceil_div_i(p.N, 128), ceil_div_i(p.M, 64), phases), b(256, 1, 1);
  switch (mode) {
    case 0: gemm_wmma<0><<<g, b, 0, st>>>(p); break;
    case 1: gemm_wmma<1><<<g, b, 0, st>>>(p); break;
    case 2: gemm_wmma<2><<<g, b, 0, st>>>(p); break;
    default: gemm_wmma<3><<<g, b, 0, st>>>(p); break;
  }
}

extern "C" void kernel_launch(void* const* d_in, const int* in_sizes, int n_in,
                              void* d_out, int out_size, void* d_ws, size_t ws_size,
                              hipStream_t stream) {
  (void)in_sizes; (void)n_in; (void)out_size; (void)ws_size;
  const float* latent   = (const float*)d_in[0];
  const float* emb_std  = (const float*)d_in[1];
  const float* emb_mean = (const float*)d_in[2];
  const float* quant_w  = (const float*)d_in[3];
  const float* quant_b  = (const float*)d_in[4];
  const float* up_w     = (const float*)d_in[5];
  const float* up_b     = (const float*)d_in[6];
  const float* n1w      = (const float*)d_in[7];
  const float* n1b      = (const float*)d_in[8];
  const float* ipw      = (const float*)d_in[9];
  const float* ipb      = (const float*)d_in[10];
  const float* opw      = (const float*)d_in[11];
  const float* opb      = (const float*)d_in[12];
  const float* ls1      = (const float*)d_in[13];
  const float* n2w      = (const float*)d_in[14];
  const float* n2b      = (const float*)d_in[15];
  const float* l1w      = (const float*)d_in[16];
  const float* l1b      = (const float*)d_in[17];
  const float* l2w      = (const float*)d_in[18];
  const float* l2b      = (const float*)d_in[19];
  const float* ls2      = (const float*)d_in[20];
  const float* tproj_w  = (const float*)d_in[21];
  const float* tproj_b  = (const float*)d_in[22];
  const float* dc0_w    = (const float*)d_in[23];
  const float* dc0_b    = (const float*)d_in[24];
  const float* df_w     = (const float*)d_in[49];
  const float* df_b     = (const float*)d_in[50];

  char* ws = (char*)d_ws;
  size_t off = 0;
  auto alloc = [&](size_t bytes) -> char* {
    char* pp = ws + off;
    off = (off + bytes + 255) & ~(size_t)255;
    return pp;
  };

  // ---- convert all weights to f16 (once per launch; deterministic) ----
  const size_t WTOT = 41460160;
  _Float16* wbase = (_Float16*)alloc(WTOT * 2);
  size_t woff = 0;
  auto cvt = [&](const float* src, size_t n) -> _Float16* {
    _Float16* d = wbase + woff; woff += n;
    cvt_f32_f16<<<ceil_div_i((long long)n, 256), 256, 0, stream>>>(src, d, (int)n);
    return d;
  };
  _Float16* quantW = cvt(quant_w, 512 * 512);
  _Float16* upW    = cvt(up_w, (size_t)512 * 512 * 4);
  _Float16* ipW    = cvt(ipw, (size_t)8 * 1536 * 512);
  _Float16* opW    = cvt(opw, (size_t)8 * 512 * 512);
  _Float16* l1W    = cvt(l1w, (size_t)8 * 2048 * 512);
  _Float16* l2W    = cvt(l2w, (size_t)8 * 512 * 2048);
  _Float16* tpW    = cvt(tproj_w, 512 * 512);
  _Float16* dc0W   = cvt(dc0_w, (size_t)1024 * 512 * 7);
  const int chs[5] = {1024, 512, 256, 128, 64};
  const int sstride[4] = {8, 6, 5, 4};
  _Float16 *tW[4], *raW[4], *rbW[4];
  const float *tB[4], *raB[4], *rbB[4];
  for (int i = 0; i < 4; ++i) {
    const int cin = chs[i], co = chs[i + 1], s = sstride[i];
    tW[i]  = cvt((const float*)d_in[25 + 6 * i + 0], (size_t)cin * co * 2 * s);
    tB[i]  = (const float*)d_in[25 + 6 * i + 1];
    raW[i] = cvt((const float*)d_in[25 + 6 * i + 2], (size_t)(co / 2) * co * 3);
    raB[i] = (const float*)d_in[25 + 6 * i + 3];
    rbW[i] = cvt((const float*)d_in[25 + 6 * i + 4], (size_t)co * (co / 2));
    rbB[i] = (const float*)d_in[25 + 6 * i + 5];
  }
  _Float16* dfW = cvt(df_w, 448);

  // ---- activation buffers ----
  const int T0 = 512, T1 = 1026;
  float* x0  = (float*)alloc((size_t)512 * 512 * 4);
  float* xq  = (float*)alloc((size_t)512 * 512 * 4);
  float* zT  = (float*)alloc((size_t)T1 * 512 * 4);
  float* lnb = (float*)alloc((size_t)T1 * 512 * 4);
  float* qkv = (float*)alloc((size_t)T1 * 1536 * 4);
  float* ao  = (float*)alloc((size_t)T1 * 512 * 4);
  float* ff1 = (float*)alloc((size_t)T1 * 2048 * 4);
  _Float16* tpT = (_Float16*)alloc((size_t)512 * T1 * 2);
  _Float16* c0  = (_Float16*)alloc((size_t)1024 * T1 * 2);
  _Float16 *sb[4], *vb[4];
  int Touts[4];
  {
    int Tin = T1;
    for (int i = 0; i < 4; ++i) {
      const int co = chs[i + 1];
      const int Tout = sstride[i] * (Tin + 1);
      Touts[i] = Tout;
      sb[i] = (_Float16*)alloc((size_t)co * Tout * 2);
      vb[i] = (_Float16*)alloc((size_t)(co / 2) * Tout * 2);
      Tin = Tout;
    }
  }

  auto GPmk = [](const void* A, const void* B, const float* bias, const float* ls,
                 const void* res, void* out, int M, int N, int K, int lda, int ldb,
                 int ks, int cout, int flags, int pstride, long long sm, long long sn) {
    GP p; p.A = A; p.B = B; p.bias = bias; p.ls = ls; p.res = res; p.out = out;
    p.M = M; p.N = N; p.K = K; p.lda = lda; p.ldb = ldb; p.ks = ks; p.cout = cout;
    p.flags = flags; p.pstride = pstride; p.out_sm = sm; p.out_sn = sn; return p;
  };

  // 1) embedding affine, quantizer 1x1 conv, stride-2 convtranspose (out -> (T,D))
  embed_k<<<1024, 256, 0, stream>>>(latent, emb_std, emb_mean, x0);
  launch_gemm(stream, 1, GPmk(quantW, x0, quant_b, nullptr, nullptr, xq,
                              512, T0, 512, 0, T0, 1, 0, FLAG_BIAS_M, 1, T0, 1), 1);
  launch_gemm(stream, 3, GPmk(upW, xq, up_b, nullptr, nullptr, zT,
                              512, T0 + 1, 1024, 0, T0, 4, 512, FLAG_BIAS_M, 2, 1, 512), 2);

  // 2) transformer (8 layers)
  for (int l = 0; l < 8; ++l) {
    ln_k<<<ceil_div_i(T1, 8), 256, 0, stream>>>(zT, n1w + l * 512, n1b + l * 512, lnb, T1);
    launch_gemm(stream, 0, GPmk(lnb, ipW + (size_t)l * 786432, ipb + l * 1536, nullptr, nullptr,
                                qkv, T1, 1536, 512, 512, 512, 1, 0, 0, 1, 1536, 1), 1);
    rope_k<<<ceil_div_i((long long)T1 * 256, 256), 256, 0, stream>>>(qkv, T1);
    attn_k<<<T1, 256, 0, stream>>>(qkv, ao, T1);
    launch_gemm(stream, 0, GPmk(ao, opW + (size_t)l * 262144, opb + l * 512, ls1 + l * 512,
                                zT, zT, T1, 512, 512, 512, 512, 1, 0, 0, 1, 512, 1), 1);
    ln_k<<<ceil_div_i(T1, 8), 256, 0, stream>>>(zT, n2w + l * 512, n2b + l * 512, lnb, T1);
    launch_gemm(stream, 0, GPmk(lnb, l1W + (size_t)l * 1048576, l1b + l * 2048, nullptr, nullptr,
                                ff1, T1, 2048, 512, 512, 512, 1, 0, FLAG_GELU, 1, 2048, 1), 1);
    launch_gemm(stream, 0, GPmk(ff1, l2W + (size_t)l * 1048576, l2b + l * 512, ls2 + l * 512,
                                zT, zT, T1, 512, 2048, 2048, 2048, 1, 0, 0, 1, 512, 1), 1);
  }

  // 3) tproj (writes (D,T) channel-major f16), dc0 conv k=7
  launch_gemm(stream, 0, GPmk(zT, tpW, tproj_b, nullptr, nullptr, tpT,
                              T1, 512, 512, 512, 512, 1, 0, FLAG_OUTHALF, 1, 1, T1), 1);
  launch_gemm(stream, 2, GPmk(dc0W, tpT, dc0_b, nullptr, nullptr, c0,
                              1024, T1, 3584, 0, T1, 7, 0,
                              FLAG_BIAS_M | FLAG_BHALF | FLAG_OUTHALF, 1, T1, 1), 1);

  // 4) upsampling stages
  const _Float16* zin = c0;
  int Tin = T1;
  for (int i = 0; i < 4; ++i) {
    const int cin = chs[i], co = chs[i + 1], s = sstride[i], Tout = Touts[i];
    const int f = FLAG_BIAS_M | FLAG_BELU | FLAG_BHALF | FLAG_OUTHALF;
    launch_gemm(stream, 3, GPmk(tW[i], zin, tB[i], nullptr, nullptr, sb[i],
                                co, Tin + 1, 2 * cin, 0, Tin, 2 * s, co, f, s, Tout, 1), s);
    launch_gemm(stream, 2, GPmk(raW[i], sb[i], raB[i], nullptr, nullptr, vb[i],
                                co / 2, Tout, co * 3, 0, Tout, 3, 0, f, 1, Tout, 1), 1);
    launch_gemm(stream, 2, GPmk(rbW[i], vb[i], rbB[i], nullptr, sb[i], sb[i],
                                co, Tout, co / 2, 0, Tout, 1, 0, f, 1, Tout, 1), 1);
    zin = sb[i]; Tin = Tout;
  }

  // 5) final conv to 1 channel (f32 output)
  launch_gemm(stream, 2, GPmk(dfW, sb[3], df_b, nullptr, nullptr, d_out,
                              1, Touts[3], 448, 0, Touts[3], 7, 0,
                              FLAG_BIAS_M | FLAG_BELU | FLAG_BHALF, 1, Touts[3], 1), 1);
}
